// DeltaUnit_42288247996736
// MI455X (gfx1250) — matrix operand order repaired and checked by reading, not compiled, not run on hardware
//
#include <hip/hip_runtime.h>

// ---------------------------------------------------------------------------
// DeltaNet forward for MI455X (gfx1250, wave32, WMMA).
// B=2, N=2048, D=1024, H=16, HD=64  ->  M = 4096 rows.
// Big GEMMs (q/k/v proj, final @Wo) run on v_wmma_f32_16x16x32_bf16 with
// GLOBAL_LOAD_ASYNC_TO_LDS staging (ASYNCcnt) + global_prefetch.
// ---------------------------------------------------------------------------

typedef __attribute__((ext_vector_type(16))) __bf16        bf16x16;
typedef __attribute__((ext_vector_type(8)))  float         f32x8;
typedef __attribute__((ext_vector_type(4)))  unsigned int  u32x4;

union Frag {            // 16 bf16 = 32 bytes = 8 VGPRs
  u32x4   u[2];
  bf16x16 v;
};

#define GM   4096
#define GD   1024
#define GH   16
#define GHD  64

// ---------------- CDNA5 async LDS-DMA helpers (inline asm) -----------------
// Flat pointer to an LDS object = {shared aperture, lds_offset[31:0]}
// (ISA section 10.2), so truncation recovers the wave-relative LDS offset the
// async instructions expect in VDST.
__device__ __forceinline__ unsigned int lds_off(const void* p) {
  return (unsigned int)(size_t)p;
}
__device__ __forceinline__ void async_load_b128(unsigned int lds, const void* g) {
  asm volatile("global_load_async_to_lds_b128 %0, %1, off"
               :: "v"(lds), "v"((unsigned long long)g) : "memory");
}
__device__ __forceinline__ void async_load_b32(unsigned int lds, const void* g) {
  asm volatile("global_load_async_to_lds_b32 %0, %1, off"
               :: "v"(lds), "v"((unsigned long long)g) : "memory");
}
__device__ __forceinline__ void wait_async0() {
  asm volatile("s_wait_asynccnt 0x0" ::: "memory");
}

__device__ __forceinline__ float fast_sigmoid(float x) {
  return __builtin_amdgcn_rcpf(1.f + __expf(-x));   // v_rcp_f32, no IEEE div
}

// ------------------------------ f32 -> bf16 --------------------------------
__global__ void cvt_f32_bf16(const float* __restrict__ src,
                             __bf16* __restrict__ dst, int n) {
  int i = blockIdx.x * blockDim.x + threadIdx.x;
  int stride = gridDim.x * blockDim.x;
  for (; i < n; i += stride) dst[i] = (__bf16)src[i];
}

// ------------------------------ WMMA GEMM ----------------------------------
// C[M,N] = act(A[M,K] @ B[K,N]); A,B bf16 row-major, C f32.
// Block tile 128x128, BK=64, 8 waves in a 4x2 grid, each wave 32x64
// (2x4 subtiles of 16x16). A tile staged by async LDS-DMA; B tile staged
// transposed so fragments are contiguous ds_load_b128s.
#define BM 128
#define BN 128
#define BK 64
#define LSTR 72   // BK + 8 bf16 pad (144 B rows -> conflict-free b128 reads)

__global__ __launch_bounds__(256) void gemm_bf16_wmma(
    const __bf16* __restrict__ A, const __bf16* __restrict__ B,
    float* __restrict__ C, int M, int N, int K, int act) {
  __shared__ __align__(16) __bf16 As[BM * LSTR];
  __shared__ __align__(16) __bf16 Bt[BN * LSTR];

  const int tid  = threadIdx.x;
  const int lane = tid & 31;
  const int wave = tid >> 5;
  const int wm   = wave >> 1;     // 0..3 -> 32-row strip
  const int wn   = wave & 1;      // 0..1 -> 64-col strip
  const int hl   = lane >> 4;     // half-wave
  const int l16  = lane & 15;

  const int rowBlock = blockIdx.y * BM;
  const int colBlock = blockIdx.x * BN;

  f32x8 acc[2][4] = {};

  // global->LDS tiling
  const int a_row = tid >> 1;            // 0..127
  const int a_col = (tid & 1) * 32;      // 0 / 32  (32 bf16 per thread)
  const int b_row = tid >> 2;            // 0..63
  const int b_col = (tid & 3) * 32;      // 0..96   (32 bf16 per thread)

  const unsigned int sa_off = lds_off(As + a_row * LSTR + a_col);

  for (int kt = 0; kt < K; kt += BK) {
    const __bf16* ga = A + (size_t)(rowBlock + a_row) * K + kt + a_col;
    const __bf16* gb = B + (size_t)(kt + b_row) * N + colBlock + b_col;

    // prefetch next K tile into GL2 while this one is consumed
    if (kt + BK < K) {
      __builtin_prefetch(ga + BK, 0, 3);
      __builtin_prefetch(gb + (size_t)BK * N, 0, 3);
    }

    // A tile: async LDS DMA (4 x b128 per thread), overlaps B transpose below
    #pragma unroll
    for (int i = 0; i < 4; ++i)
      async_load_b128(sa_off + i * 16, ga + i * 8);

    // B tile: load to regs, store transposed into LDS: Bt[col][k]
    #pragma unroll
    for (int i = 0; i < 4; ++i) {
      u32x4 d = *reinterpret_cast<const u32x4*>(gb + i * 8);
      const __bf16* e = reinterpret_cast<const __bf16*>(&d);
      #pragma unroll
      for (int j = 0; j < 8; ++j)
        Bt[(b_col + i * 8 + j) * LSTR + b_row] = e[j];
    }
    wait_async0();
    __syncthreads();

    #pragma unroll
    for (int s = 0; s < BK; s += 32) {
      Frag af[2], bfr[4];
      // A frag (16x32 bf16): lanes0-15 row=l16, K = s+{0..7, 16..23};
      //                      lanes16-31          K = s+{8..15, 24..31}
      #pragma unroll
      for (int i = 0; i < 2; ++i) {
        const __bf16* pa = As + (wm * 32 + i * 16 + l16) * LSTR + s + hl * 8;
        af[i].u[0] = *reinterpret_cast<const u32x4*>(pa);
        af[i].u[1] = *reinterpret_cast<const u32x4*>(pa + 16);
      }
      // B frag (32x16 bf16): lane -> col l16; K = s+{0..7,16..23} (+8 for hl=1)
      #pragma unroll
      for (int j = 0; j < 4; ++j) {
        const __bf16* pb = Bt + (wn * 64 + j * 16 + l16) * LSTR + s + hl * 8;
        bfr[j].u[0] = *reinterpret_cast<const u32x4*>(pb);
        bfr[j].u[1] = *reinterpret_cast<const u32x4*>(pb + 16);
      }
      #pragma unroll
      for (int i = 0; i < 2; ++i)
        #pragma unroll
        for (int j = 0; j < 4; ++j)
          acc[i][j] = __builtin_amdgcn_wmma_f32_16x16x32_bf16(
              false, af[i].v, false, bfr[j].v, (short)0, acc[i][j],
              false, false);
    }
    __syncthreads();
  }

  // epilogue: C/D layout -> VGPR r: lanes0-15 M=r, lanes16-31 M=8+r
  #pragma unroll
  for (int i = 0; i < 2; ++i)
    #pragma unroll
    for (int j = 0; j < 4; ++j)
      #pragma unroll
      for (int r = 0; r < 8; ++r) {
        int row = rowBlock + wm * 32 + i * 16 + hl * 8 + r;
        int col = colBlock + wn * 64 + j * 16 + l16;
        float vv = acc[i][j][r];
        if (act == 1) vv = vv * fast_sigmoid(vv);   // SiLU
        C[(size_t)row * N + col] = vv;
      }
}

// --------------------- skinny projections (N=16,16,64) ---------------------
// gamma = -sigmoid(x@Wgamma), logf = log_sigmoid(x@Wf), g1 = x@Wg1
__global__ __launch_bounds__(128) void small_proj(
    const float* __restrict__ x, const float* __restrict__ Wgamma,
    const float* __restrict__ Wf, const float* __restrict__ Wg1,
    float* __restrict__ gamma, float* __restrict__ logf_o,
    float* __restrict__ g1) {
  __shared__ float xs[GD];
  const int row = blockIdx.x;
  const int tid = threadIdx.x;
  for (int i = tid; i < GD; i += 128) xs[i] = x[(size_t)row * GD + i];
  __syncthreads();
  if (tid < 96) {
    float acc = 0.f;
    if (tid < 16) {
      for (int k2 = 0; k2 < GD; ++k2) acc += xs[k2] * Wgamma[k2 * GH + tid];
      gamma[row * GH + tid] = -fast_sigmoid(acc);
    } else if (tid < 32) {
      int c = tid - 16;
      for (int k2 = 0; k2 < GD; ++k2) acc += xs[k2] * Wf[k2 * GH + c];
      float m = fminf(acc, 0.f);                 // stable log_sigmoid
      logf_o[row * GH + c] = m - log1pf(__expf(-fabsf(acc)));
    } else {
      int c = tid - 32;
      for (int k2 = 0; k2 < GD; ++k2) acc += xs[k2] * Wg1[k2 * GHD + c];
      g1[row * GHD + c] = acc;
    }
  }
}

// ------------------- l2norm(k) in place and a = k*gamma*exp(gk) ------------
__global__ __launch_bounds__(256) void norm_a_kernel(
    float* __restrict__ k, const float* __restrict__ logf,
    const float* __restrict__ gamma, float* __restrict__ a) {
  int idx = blockIdx.x * blockDim.x + threadIdx.x;   // row*16 + h
  if (idx >= GM * GH) return;
  int row = idx >> 4, h = idx & 15;
  size_t base = (size_t)row * GD + h * GHD;
  float kv[GHD];
  float ss = 0.f;
  #pragma unroll
  for (int j = 0; j < GHD; ++j) { kv[j] = k[base + j]; ss += kv[j] * kv[j]; }
  float inv = 1.f / fmaxf(sqrtf(ss), 1e-12f);
  float scale = gamma[idx] * __expf(logf[idx]);
  #pragma unroll
  for (int j = 0; j < GHD; ++j) {
    float kn = kv[j] * inv;
    k[base + j] = kn;
    a[base + j] = kn * scale;
  }
}

// ------------------------- sequential delta-rule scan ----------------------
// One block of 64 threads per (b,h); thread owns state column S[:, tid].
// Chunk staging uses async LDS DMA (b128 granules).
#define CHUNK 32
__global__ __launch_bounds__(64) void scan_kernel(
    const float* __restrict__ q, const float* __restrict__ k,
    const float* __restrict__ v, const float* __restrict__ a,
    const float* __restrict__ logf, float* __restrict__ o) {
  __shared__ __align__(16) float qs[CHUNK][GHD], ks[CHUNK][GHD],
                                 vs[CHUNK][GHD], as_[CHUNK][GHD];
  __shared__ float dec[CHUNK];
  const int b = blockIdx.x >> 4, h = blockIdx.x & 15;
  const int tid = threadIdx.x;
  const int rbase = b * 2048;
  const int cb = h * GHD;
  float S[GHD];
  #pragma unroll
  for (int j = 0; j < GHD; ++j) S[j] = 0.f;

  for (int c0 = 0; c0 < 2048; c0 += CHUNK) {
    // 32 rows x 256B per array; thread t copies 16B granules t, t+64, ...
    #pragma unroll
    for (int i = 0; i < 8; ++i) {
      int c = tid + i * 64;            // 0..511
      int r = c >> 4, w = (c & 15) * 4;
      size_t g = (size_t)(rbase + c0 + r) * GD + cb + w;
      async_load_b128(lds_off(&qs[r][w]),  q + g);
      async_load_b128(lds_off(&ks[r][w]),  k + g);
      async_load_b128(lds_off(&vs[r][w]),  v + g);
      async_load_b128(lds_off(&as_[r][w]), a + g);
    }
    if (tid < CHUNK)
      dec[tid] = logf[(size_t)(rbase + c0 + tid) * GH + h];
    wait_async0();
    __syncthreads();
    for (int r = 0; r < CHUNK; ++r) {
      float d = __expf(dec[r]);          // uniform per-head decay
      float u = vs[r][tid];
      #pragma unroll
      for (int j = 0; j < GHD; ++j) {    // S' = d*S ; u = a.S' + v
        float sp = S[j] * d; S[j] = sp; u += as_[r][j] * sp;
      }
      float ov = 0.f;
      #pragma unroll
      for (int j = 0; j < GHD; ++j) {    // S = S' + k (x) u ; o = q.S
        float s2 = S[j] + ks[r][j] * u; S[j] = s2; ov += qs[r][j] * s2;
      }
      o[(size_t)(rbase + c0 + r) * GD + cb + tid] = ov;
    }
    __syncthreads();
  }
}

// --------- gate = sigmoid(g1@Wg2), out *= gate, LayerNorm, -> bf16 ---------
__global__ __launch_bounds__(256) void gate_ln_kernel(
    const float* __restrict__ o, const float* __restrict__ g1,
    const float* __restrict__ Wg2, const float* __restrict__ norm_w,
    __bf16* __restrict__ ln_bf) {
  __shared__ float g1s[GHD];
  __shared__ float red[256];
  const int row = blockIdx.x;
  const int tid = threadIdx.x;
  if (tid < GHD) g1s[tid] = g1[(size_t)row * GHD + tid];
  __syncthreads();
  float val[4];
  float lsum = 0.f;
  #pragma unroll
  for (int i = 0; i < 4; ++i) {
    int c = tid + i * 256;
    float accd = 0.f;
    for (int k2 = 0; k2 < GHD; ++k2) accd += g1s[k2] * Wg2[k2 * GD + c];
    val[i] = o[(size_t)row * GD + c] * fast_sigmoid(accd);
    lsum += val[i];
  }
  red[tid] = lsum; __syncthreads();
  for (int s = 128; s > 0; s >>= 1) {
    if (tid < s) red[tid] += red[tid + s];
    __syncthreads();
  }
  float mean = red[0] * (1.f / GD);
  __syncthreads();
  float vsum = 0.f;
  #pragma unroll
  for (int i = 0; i < 4; ++i) { float dd = val[i] - mean; vsum += dd * dd; }
  red[tid] = vsum; __syncthreads();
  for (int s = 128; s > 0; s >>= 1) {
    if (tid < s) red[tid] += red[tid + s];
    __syncthreads();
  }
  float inv = rsqrtf(red[0] * (1.f / GD) + 1e-5f);
  #pragma unroll
  for (int i = 0; i < 4; ++i) {
    int c = tid + i * 256;
    float y = (val[i] - mean) * inv * norm_w[c];
    ln_bf[(size_t)row * GD + c] = (__bf16)y;
  }
}

// ------------------------------- launcher ----------------------------------
extern "C" void kernel_launch(void* const* d_in, const int* in_sizes, int n_in,
                              void* d_out, int out_size, void* d_ws,
                              size_t ws_size, hipStream_t stream) {
  (void)in_sizes; (void)n_in; (void)out_size; (void)ws_size;
  const float* x      = (const float*)d_in[0];
  const float* Wq     = (const float*)d_in[1];
  const float* Wk     = (const float*)d_in[2];
  const float* Wv     = (const float*)d_in[3];
  const float* Wgamma = (const float*)d_in[4];
  const float* Wf     = (const float*)d_in[5];
  const float* Wg1    = (const float*)d_in[6];
  const float* Wg2    = (const float*)d_in[7];
  const float* Wo     = (const float*)d_in[8];
  const float* norm_w = (const float*)d_in[9];
  float* out = (float*)d_out;

  char* p = (char*)d_ws;
  auto alloc = [&](size_t bytes) -> void* {
    void* r = (void*)p;
    p += (bytes + 255) & ~(size_t)255;
    return r;
  };
  __bf16* x_bf  = (__bf16*)alloc((size_t)GM * GD * 2);
  __bf16* Wq_bf = (__bf16*)alloc((size_t)GD * GD * 2);
  __bf16* Wk_bf = (__bf16*)alloc((size_t)GD * GD * 2);
  __bf16* Wv_bf = (__bf16*)alloc((size_t)GD * GD * 2);
  __bf16* Wo_bf = (__bf16*)alloc((size_t)GD * GD * 2);
  float*  qb    = (float*)alloc((size_t)GM * GD * 4);
  float*  kb    = (float*)alloc((size_t)GM * GD * 4);
  float*  vb    = (float*)alloc((size_t)GM * GD * 4);
  float*  ab    = (float*)alloc((size_t)GM * GD * 4);
  float*  ob    = (float*)alloc((size_t)GM * GD * 4);
  float*  logf_b  = (float*)alloc((size_t)GM * GH * 4);
  float*  gamma_b = (float*)alloc((size_t)GM * GH * 4);
  float*  g1_b  = (float*)alloc((size_t)GM * GHD * 4);
  __bf16* ln_bf = (__bf16*)alloc((size_t)GM * GD * 2);

  // 1) bf16 conversions
  cvt_f32_bf16<<<1024, 256, 0, stream>>>(x,  x_bf,  GM * GD);
  cvt_f32_bf16<<<1024, 256, 0, stream>>>(Wq, Wq_bf, GD * GD);
  cvt_f32_bf16<<<1024, 256, 0, stream>>>(Wk, Wk_bf, GD * GD);
  cvt_f32_bf16<<<1024, 256, 0, stream>>>(Wv, Wv_bf, GD * GD);
  cvt_f32_bf16<<<1024, 256, 0, stream>>>(Wo, Wo_bf, GD * GD);

  // 2) big WMMA GEMMs: q/k/v = silu(x @ W*)
  dim3 ggrid(GD / BN, GM / BM);   // (8, 32)
  gemm_bf16_wmma<<<ggrid, 256, 0, stream>>>(x_bf, Wq_bf, qb, GM, GD, GD, 1);
  gemm_bf16_wmma<<<ggrid, 256, 0, stream>>>(x_bf, Wk_bf, kb, GM, GD, GD, 1);
  gemm_bf16_wmma<<<ggrid, 256, 0, stream>>>(x_bf, Wv_bf, vb, GM, GD, GD, 1);

  // 3) skinny projections
  small_proj<<<GM, 128, 0, stream>>>(x, Wgamma, Wf, Wg1,
                                     gamma_b, logf_b, g1_b);

  // 4) l2norm(k), a = k * gamma * exp(logf)
  norm_a_kernel<<<(GM * GH + 255) / 256, 256, 0, stream>>>(kb, logf_b,
                                                           gamma_b, ab);

  // 5) sequential delta-rule scan: 32 blocks (B*H), 64 threads each
  scan_kernel<<<32, 64, 0, stream>>>(qb, kb, vb, ab, logf_b, ob);

  // 6) gating + LayerNorm -> bf16
  gate_ln_kernel<<<GM, 256, 0, stream>>>(ob, g1_b, Wg2, norm_w, ln_bf);

  // 7) final WMMA GEMM: out = ln @ Wo
  gemm_bf16_wmma<<<ggrid, 256, 0, stream>>>(ln_bf, Wo_bf, out, GM, GD, GD, 0);
}
